// PointConvAttention_76175539962313
// MI455X (gfx1250) — matrix operand order repaired and checked by reading, not compiled, not run on hardware
//
#include <hip/hip_runtime.h>

// ---------------- types ----------------
typedef __attribute__((ext_vector_type(16))) __bf16 v16bf;
typedef __attribute__((ext_vector_type(8)))  float  v8f;
typedef __attribute__((ext_vector_type(4)))  int    v4i;

union Frag {
  v16bf v;
  unsigned short u[16];
  uint4 q[2];
};

// ---------------- constants ----------------
#define NPTS  16384
#define BATCH 8
#define CIN   64
#define KNN   8
#define WAVES 4                    // waves (16-point tiles) per block
#define TPB   (WAVES * 32)

// weight-fragment arena offsets in halfwords; one fragment = 512 bf16 (32 lanes x 16)
// fragments stored kt-major within a (sub)layer: frag = kt*NM + mt
// 256-output layers are split into two 128-row halves so every GEMM has NM<=8.
constexpr int FR_WM0   = 0;                         // 128x512 : NM=8,  NK=16
constexpr int FR_WM1   = FR_WM0 + 8 * 16 * 512;     // 128x128 : NM=8,  NK=4
constexpr int FR_WM2   = FR_WM1 + 8 * 4 * 512;      // 8->16x128: NM=1, NK=4
constexpr int FR_M0A   = FR_WM2 + 1 * 4 * 512;      // rows 0..127 of 256x576
constexpr int FR_M0B   = FR_M0A + 8 * 18 * 512;     // rows 128..255 of 256x576
constexpr int FR_M1A   = FR_M0B + 8 * 18 * 512;     // rows 0..127 of 256x256
constexpr int FR_M1B   = FR_M1A + 8 * 8 * 512;      // rows 128..255 of 256x256
constexpr int FR_M2    = FR_M1B + 8 * 8 * 512;      // 128x256 : NM=8,  NK=8
constexpr int FR_TOTAL = FR_M2 + 8 * 8 * 512;       // 329,728 halfwords = 659,456 B

// ---------------- async global->LDS copy (CDNA5 ASYNCcnt path) ----------------
#if __has_builtin(__builtin_amdgcn_global_load_async_to_lds_b128)
#define HAVE_ASYNC_LDS 1
__device__ inline void async_copy16(const void* g, void* l) {
  __builtin_amdgcn_global_load_async_to_lds_b128(
      (__attribute__((address_space(1))) v4i*)(unsigned long long)(uintptr_t)g,
      (__attribute__((address_space(3))) v4i*)(unsigned)(uintptr_t)l, 0, 0);
}
__device__ inline void async_wait0() {
#if __has_builtin(__builtin_amdgcn_s_wait_asynccnt)
  __builtin_amdgcn_s_wait_asynccnt(0);
#else
  asm volatile("s_wait_asynccnt 0x0" ::: "memory");
#endif
}
#else
#define HAVE_ASYNC_LDS 0
#endif

// ---------------- helpers ----------------
__device__ __host__ inline unsigned short f2bf(float f) {
  union { float f; unsigned u; } cv; cv.f = f;
  unsigned u = cv.u;
  return (unsigned short)((u + 0x7FFFu + ((u >> 16) & 1u)) >> 16);
}
__device__ inline float bf2f(unsigned short h) {
  union { unsigned u; float f; } cv; cv.u = ((unsigned)h) << 16;
  return cv.f;
}
__device__ inline v8f vzero() {
  v8f z;
#pragma unroll
  for (int i = 0; i < 8; ++i) z[i] = 0.f;
  return z;
}
__device__ inline v8f wmma_bf16(const Frag& a, const Frag& b, v8f c) {
  return __builtin_amdgcn_wmma_f32_16x16x32_bf16(
      /*neg_a=*/false, a.v, /*neg_b=*/false, b.v,
      /*c_mod=*/(short)0, c, /*reuse_a=*/false, /*reuse_b=*/false);
}
__device__ inline void load_frag(const unsigned short* base, int lane, Frag& f) {
  const uint4* p = (const uint4*)(base + lane * 16);
  f.q[0] = p[0];
  f.q[1] = p[1];
}
__device__ inline void biasact(v8f& c, const float* bias, int mbase, int cout,
                               bool relu, int grp) {
#pragma unroll
  for (int v = 0; v < 8; ++v) {
    int m = mbase + v + 8 * grp;
    float x = c[v] + (m < cout ? bias[m] : 0.f);
    c[v] = relu ? fmaxf(x, 0.f) : x;
  }
}
// write one 16x16 f32 accumulator tile to LDS in bf16 B-fragment layout
__device__ inline void store_act(unsigned short* buf, int chbase, const v8f& c,
                                 int grp, int pl) {
  int ch0 = chbase + 8 * grp;  // 8 consecutive channels land in one 16B run
  int addr = (ch0 >> 5) * 512 + ((((ch0 >> 4) & 1) * 16 + pl) * 16) + (ch0 & 15);
  unsigned t[4];
#pragma unroll
  for (int i = 0; i < 4; ++i) {
    unsigned lo = f2bf(c[2 * i]);
    unsigned hi = f2bf(c[2 * i + 1]);
    t[i] = lo | (hi << 16);
  }
  uint4 q; q.x = t[0]; q.y = t[1]; q.z = t[2]; q.w = t[3];
  *(uint4*)(buf + addr) = q;
}

// Block-cooperative, double-buffered, software-pipelined GEMM layer (NM<=8).
// A fragments streamed global -> shared LDS stage (async DMA when available);
// B fragments are per-wave LDS; NM m-tiles accumulated in registers.
// kt loop fully unrolled: the acc chain stays a straight SSA D->C chain so the
// allocator keeps accumulators in place (no phi copies / hazard NOPs).
template <int NM, int NK>
__device__ inline void gemm_staged(const unsigned short* __restrict__ abase,
                                   const unsigned short* __restrict__ bsrc,
                                   unsigned short* __restrict__ stage0,
                                   unsigned short* __restrict__ stage1,
                                   int lane, int wave, v8f* acc) {
  constexpr int ITERS = (NM >= 2) ? (NM / 2) : 1;  // chunks per lane per k-tile
  constexpr int MASK = NM * 64 - 1;                // NM is a power of two
#pragma unroll
  for (int t = 0; t < NM; ++t) acc[t] = vzero();

#if HAVE_ASYNC_LDS
  auto stage_kt = [&](int kt, unsigned short* dst) {
    const unsigned short* src = abase + kt * NM * 512;
#pragma unroll
    for (int i = 0; i < ITERS; ++i) {
      int j = (wave * (NM * 16) + i * 32 + lane) & MASK;
      async_copy16(src + j * 8, dst + j * 8);
    }
  };
  stage_kt(0, stage0);
  async_wait0();
  __syncthreads();
#pragma unroll
  for (int kt = 0; kt < NK; ++kt) {
    unsigned short* scur = (kt & 1) ? stage1 : stage0;
    unsigned short* snxt = (kt & 1) ? stage0 : stage1;
    if (kt + 1 < NK) stage_kt(kt + 1, snxt);  // overlaps with WMMA below
    Frag bm;
    load_frag(bsrc + kt * 512, lane, bm);
#pragma unroll
    for (int mt = 0; mt < NM; ++mt) {
      Frag a;
      load_frag(scur + mt * 512, lane, a);
      acc[mt] = wmma_bf16(a, bm, acc[mt]);
    }
    async_wait0();
    __syncthreads();
  }
#else
  // fallback: bounce through registers (divergence-free, no guards)
  uint4 regs[ITERS];
  {
    const uint4* src = (const uint4*)abase;
#pragma unroll
    for (int i = 0; i < ITERS; ++i)
      regs[i] = src[(wave * (NM * 16) + i * 32 + lane) & MASK];
#pragma unroll
    for (int i = 0; i < ITERS; ++i)
      ((uint4*)stage0)[(wave * (NM * 16) + i * 32 + lane) & MASK] = regs[i];
  }
  __syncthreads();
#pragma unroll
  for (int kt = 0; kt < NK; ++kt) {
    unsigned short* scur = (kt & 1) ? stage1 : stage0;
    unsigned short* snxt = (kt & 1) ? stage0 : stage1;
    if (kt + 1 < NK) {
      const uint4* src = (const uint4*)(abase + (kt + 1) * NM * 512);
#pragma unroll
      for (int i = 0; i < ITERS; ++i)
        regs[i] = src[(wave * (NM * 16) + i * 32 + lane) & MASK];
    }
    Frag bm;
    load_frag(bsrc + kt * 512, lane, bm);
#pragma unroll
    for (int mt = 0; mt < NM; ++mt) {
      Frag a;
      load_frag(scur + mt * 512, lane, a);
      acc[mt] = wmma_bf16(a, bm, acc[mt]);
    }
    if (kt + 1 < NK) {
#pragma unroll
      for (int i = 0; i < ITERS; ++i)
        ((uint4*)snxt)[(wave * (NM * 16) + i * 32 + lane) & MASK] = regs[i];
    }
    __syncthreads();
  }
#endif
}

// ---------------- prep kernel: f32 [Cout,Cin] -> bf16 A-fragments ----------------
// A 16-bit 16x32 layout: lanes 0-15 K=0..15, lanes 16-31 K=16..31
// (contiguous, matching B); fragments kt-major within (sub)layer.
__global__ void prep_weights(const float* __restrict__ W, int cout, int cin,
                             unsigned short* __restrict__ dst, int nElems) {
  int nm = (cout + 15) / 16;
  for (int e = blockIdx.x * blockDim.x + threadIdx.x; e < nElems;
       e += gridDim.x * blockDim.x) {
    int frag = e >> 9;
    int q = e & 511;
    int kt = frag / nm;
    int mt = frag % nm;
    int lanef = q >> 4;
    int h = q & 15;
    int m_local = lanef & 15;
    int grpf = lanef >> 4;
    int v = h >> 1;
    int half = h & 1;
    int k_local = grpf * 16 + h;
    int m = mt * 16 + m_local;
    int k = kt * 32 + k_local;
    float val = (m < cout) ? W[m * cin + k] : 0.f;
    dst[e] = f2bf(val);
  }
}

// ---------------- main kernel ----------------
struct alignas(16) WaveMem {
  unsigned short fragbuf[18 * 512];  // 18 KB: layer input fragments (B layout)
  unsigned short actbuf[8 * 512];    //  8 KB: intermediate activations
  float wbuf[16 * 8];                // softmax attention weights [point][k]
  int idxbuf[16 * 8];                // neighbor indices [point][k]
};
struct alignas(16) BlockMem {
  WaveMem w[WAVES];                  // 4 x 27 KB
  unsigned short stage[2][8 * 512];  // 2 x 8 KB shared weight staging (NM<=8)
};

__global__ __launch_bounds__(TPB) void pca_main(
    const float* __restrict__ feature, const int* __restrict__ idx,
    const float* __restrict__ append, const float* __restrict__ bias_wm0,
    const float* __restrict__ bias_wm1, const float* __restrict__ bias_wm2,
    const float* __restrict__ bias_m0, const float* __restrict__ bias_m1,
    const float* __restrict__ bias_m2, const unsigned short* __restrict__ wfrag,
    float* __restrict__ out) {
  __shared__ BlockMem smem;
  const int tid = threadIdx.x;
  const int wave = tid >> 5;
  const int lane = tid & 31;
  WaveMem& sm = smem.w[wave];
  unsigned short* st0 = smem.stage[0];
  unsigned short* st1 = smem.stage[1];
  const int grp = lane >> 4;  // half-wave group
  const int pl = lane & 15;   // point within tile

  const int blocksPerB = NPTS / (16 * WAVES);
  const int b = blockIdx.x / blocksPerB;
  const int nb = (blockIdx.x % blocksPerB) * (16 * WAVES) + wave * 16;

  // ---- stage 0: neighbor indices for this wave's 16 points ----
  for (int e = lane; e < 16 * KNN; e += 32)
    sm.idxbuf[e] = idx[(b * NPTS + nb + (e >> 3)) * KNN + (e & 7)];

  // ---- stage 1: gather grouped features (512 ch x 16 pts) -> fragbuf bf16 ----
  for (int e = lane; e < 512 * 16; e += 32) {
    int p = e >> 9, k = (e >> 6) & 7, c = e & 63;
    int id = sm.idxbuf[p * KNN + k];
    float val = feature[(b * CIN + c) * NPTS + id];
    int ch = k * 64 + c;
    int addr = (ch >> 5) * 512 + ((((ch >> 4) & 1) * 16 + p) * 16) + (ch & 15);
    sm.fragbuf[addr] = f2bf(val);
  }
  // ---- stage 2: append features -> fragments 16..17 (channels 512..575) ----
  for (int e = lane; e < 64 * 16; e += 32) {
    int p = e & 15, c = e >> 4;
    float val = append[(b * 64 + c) * NPTS + nb + p];
    int ch = 512 + c;
    int addr = (ch >> 5) * 512 + ((((ch >> 4) & 1) * 16 + p) * 16) + (ch & 15);
    sm.fragbuf[addr] = f2bf(val);
  }
  __syncthreads();

  v8f acc[8];

  // ======== weight_mlp L0: 512 -> 128 (fragbuf -> actbuf frags 0..3) ========
  gemm_staged<8, 16>(wfrag + FR_WM0, sm.fragbuf, st0, st1, lane, wave, acc);
#pragma unroll
  for (int mt = 0; mt < 8; ++mt) {
    biasact(acc[mt], bias_wm0, mt * 16, 128, true, grp);
    store_act(sm.actbuf, mt * 16, acc[mt], grp, pl);
  }

  // ======== weight_mlp L1: 128 -> 128 (actbuf 0..3 -> actbuf 4..7) ========
  gemm_staged<8, 4>(wfrag + FR_WM1, sm.actbuf, st0, st1, lane, wave, acc);
#pragma unroll
  for (int mt = 0; mt < 8; ++mt) {
    biasact(acc[mt], bias_wm1, mt * 16, 128, true, grp);
    store_act(sm.actbuf, 128 + mt * 16, acc[mt], grp, pl);
  }

  // ======== weight_mlp L2: 128 -> 8 (padded to 16) + softmax over K ========
  gemm_staged<1, 4>(wfrag + FR_WM2, sm.actbuf + 4 * 512, st0, st1, lane, wave, acc);
  v8f lg = acc[0];
  biasact(lg, bias_wm2, 0, 8, false, grp);
  // C-tile layout: lanes 0..15 hold point pl; neighbor k = accumulator index
  float mx = lg[0];
#pragma unroll
  for (int v = 1; v < 8; ++v) mx = fmaxf(mx, lg[v]);
  float ex[8];
  float s = 0.f;
#pragma unroll
  for (int v = 0; v < 8; ++v) { ex[v] = __expf(lg[v] - mx); s += ex[v]; }
  float inv = 1.f / s;
  if (grp == 0) {
#pragma unroll
    for (int v = 0; v < 8; ++v) sm.wbuf[pl * 8 + v] = ex[v] * inv;
  }

  // ======== attention rescale of grouped features in fragbuf ========
  // each lane's 32B chunk = 16 channels of one neighbor k for point pl
  for (int f = 0; f < 16; ++f) {
    int k = (f * 32 + grp * 16) >> 6;
    float wv = sm.wbuf[pl * 8 + k];
    uint4* p = (uint4*)(sm.fragbuf + f * 512 + lane * 16);
#pragma unroll
    for (int half = 0; half < 2; ++half) {
      uint4 q = p[half];
      unsigned qq[4] = {q.x, q.y, q.z, q.w};
#pragma unroll
      for (int i = 0; i < 4; ++i) {
        unsigned lo = qq[i] & 0xFFFFu, hi = qq[i] >> 16;
        unsigned nlo = f2bf(bf2f((unsigned short)lo) * wv);
        unsigned nhi = f2bf(bf2f((unsigned short)hi) * wv);
        qq[i] = nlo | (nhi << 16);
      }
      q.x = qq[0]; q.y = qq[1]; q.z = qq[2]; q.w = qq[3];
      p[half] = q;
    }
  }
  __syncthreads();

  // ======== mlp L0 (rows 0..127): 576 -> 128 (fragbuf -> actbuf 0..3) ========
  gemm_staged<8, 18>(wfrag + FR_M0A, sm.fragbuf, st0, st1, lane, wave, acc);
#pragma unroll
  for (int mt = 0; mt < 8; ++mt) {
    biasact(acc[mt], bias_m0, mt * 16, 128, true, grp);
    store_act(sm.actbuf, mt * 16, acc[mt], grp, pl);
  }
  // ======== mlp L0 (rows 128..255): 576 -> 128 (fragbuf -> actbuf 4..7) ========
  gemm_staged<8, 18>(wfrag + FR_M0B, sm.fragbuf, st0, st1, lane, wave, acc);
#pragma unroll
  for (int mt = 0; mt < 8; ++mt) {
    biasact(acc[mt], bias_m0 + 128, mt * 16, 128, true, grp);
    store_act(sm.actbuf, 128 + mt * 16, acc[mt], grp, pl);
  }

  // ======== mlp L1 (rows 0..127): 256 -> 128 (actbuf -> fragbuf 0..3) ========
  gemm_staged<8, 8>(wfrag + FR_M1A, sm.actbuf, st0, st1, lane, wave, acc);
#pragma unroll
  for (int mt = 0; mt < 8; ++mt) {
    biasact(acc[mt], bias_m1, mt * 16, 128, true, grp);
    store_act(sm.fragbuf, mt * 16, acc[mt], grp, pl);
  }
  // ======== mlp L1 (rows 128..255): 256 -> 128 (actbuf -> fragbuf 4..7) ========
  gemm_staged<8, 8>(wfrag + FR_M1B, sm.actbuf, st0, st1, lane, wave, acc);
#pragma unroll
  for (int mt = 0; mt < 8; ++mt) {
    biasact(acc[mt], bias_m1 + 128, mt * 16, 128, true, grp);
    store_act(sm.fragbuf, 128 + mt * 16, acc[mt], grp, pl);
  }

  // ======== mlp L2: 256 -> 128 -> global ========
  gemm_staged<8, 8>(wfrag + FR_M2, sm.fragbuf, st0, st1, lane, wave, acc);
#pragma unroll
  for (int mt = 0; mt < 8; ++mt) {
    biasact(acc[mt], bias_m2, mt * 16, 128, false, grp);
#pragma unroll
    for (int v = 0; v < 8; ++v) {
      int m = mt * 16 + v + 8 * grp;
      out[(long)(b * 128 + m) * NPTS + nb + pl] = acc[mt][v];
    }
  }
}

// ---------------- host launcher ----------------
extern "C" void kernel_launch(void* const* d_in, const int* in_sizes, int n_in,
                              void* d_out, int out_size, void* d_ws, size_t ws_size,
                              hipStream_t stream) {
  const float* feature = (const float*)d_in[0];
  const int* idx       = (const int*)d_in[1];
  const float* append  = (const float*)d_in[2];
  const float* W_wm0 = (const float*)d_in[3];
  const float* b_wm0 = (const float*)d_in[4];
  const float* W_wm1 = (const float*)d_in[5];
  const float* b_wm1 = (const float*)d_in[6];
  const float* W_wm2 = (const float*)d_in[7];
  const float* b_wm2 = (const float*)d_in[8];
  const float* W_m0  = (const float*)d_in[9];
  const float* b_m0  = (const float*)d_in[10];
  const float* W_m1  = (const float*)d_in[11];
  const float* b_m1  = (const float*)d_in[12];
  const float* W_m2  = (const float*)d_in[13];
  const float* b_m2  = (const float*)d_in[14];

  if (ws_size < (size_t)FR_TOTAL * sizeof(unsigned short)) return;
  unsigned short* wfrag = (unsigned short*)d_ws;

  auto prep_launch = [&](const float* W, int cout, int cin, int off) {
    int nm = (cout + 15) / 16, nk = cin / 32;
    int n = nm * nk * 512;
    prep_weights<<<(n + 255) / 256, 256, 0, stream>>>(W, cout, cin, wfrag + off, n);
  };
  prep_launch(W_wm0, 128, 512, FR_WM0);
  prep_launch(W_wm1, 128, 128, FR_WM1);
  prep_launch(W_wm2, 8, 128, FR_WM2);
  prep_launch(W_m0,             128, 576, FR_M0A);   // rows 0..127
  prep_launch(W_m0 + 128 * 576, 128, 576, FR_M0B);   // rows 128..255
  prep_launch(W_m1,             128, 256, FR_M1A);   // rows 0..127
  prep_launch(W_m1 + 128 * 256, 128, 256, FR_M1B);   // rows 128..255
  prep_launch(W_m2, 128, 256, FR_M2);

  pca_main<<<BATCH * (NPTS / (16 * WAVES)), TPB, 0, stream>>>(
      feature, idx, append, b_wm0, b_wm1, b_wm2, b_m0, b_m1, b_m2, wfrag,
      (float*)d_out);
}
